// approx_Linear_int8_5239860101336
// MI455X (gfx1250) — 8-bit matrix operand order repaired and checked by reading, compiled, not run on hardware
//
#include <hip/hip_runtime.h>


#define B_ROWS 1024
#define IN_K   512
#define OUT_N  512

#define OFF_AMAX 0
#define OFF_SX   1024
#define OFF_SW   5120
#define OFF_R    8192
#define OFF_XA   270336
#define OFF_WA   794624

#define LDS_R_BYTES   262144
#define LDS_XA_BYTES  32768
#define LDS_WA_BYTES  16384
#define LDS_TOTAL     (LDS_R_BYTES + LDS_XA_BYTES + LDS_WA_BYTES)

typedef int v8i __attribute__((ext_vector_type(8)));
typedef unsigned v4u __attribute__((ext_vector_type(4)));
typedef unsigned v8u __attribute__((ext_vector_type(8)));


__device__ __forceinline__ void async_copy_b128(void* lds_ptr, const void* gptr) {
  unsigned loff = (unsigned)(unsigned long long)lds_ptr;
  asm volatile("global_load_async_to_lds_b128 %0, %1, off"
               :: "v"(loff), "v"(gptr) : "memory");
}

__device__ __forceinline__ void wait_asynccnt0() {
  asm volatile("s_wait_asynccnt 0x0" ::: "memory");
}

__device__ __forceinline__ void tdm_load_2d_f32(void* lds_ptr, const void* gptr,
                                                unsigned rows, unsigned cols) {
  unsigned long long ga = (unsigned long long)gptr;
  unsigned lds_addr = (unsigned)(unsigned long long)lds_ptr;
  v4u g0;
  g0[0] = 1u;
  g0[1] = lds_addr;
  g0[2] = (unsigned)ga;
  g0[3] = (unsigned)(ga >> 32) | (2u << 30);
  v8u g1;
  g1[0] = 2u << 16;
  g1[1] = cols << 16;
  g1[2] = rows << 16;
  g1[3] = cols << 16;
  g1[4] = rows;
  g1[5] = cols;
  g1[6] = 0u;
  g1[7] = 0u;
  asm volatile("tensor_load_to_lds %0, %1" :: "s"(g0), "s"(g1) : "memory");
}

__device__ __forceinline__ void wait_tensorcnt0() {
  __builtin_amdgcn_s_wait_tensorcnt(0);
}


__global__ void init_kernel(unsigned* amax) {
  if (threadIdx.x < 2) amax[threadIdx.x] = 0u;
}

__global__ void absmax_kernel(const float* __restrict__ src, int n, unsigned* __restrict__ slot) {
  __shared__ float sm[256];
  float m = 0.0f;
  for (int i = blockIdx.x * blockDim.x + threadIdx.x; i < n; i += gridDim.x * blockDim.x)
    m = fmaxf(m, fabsf(src[i]));
  sm[threadIdx.x] = m;
  __syncthreads();
  for (int s = 128; s > 0; s >>= 1) {
    if (threadIdx.x < (unsigned)s) sm[threadIdx.x] = fmaxf(sm[threadIdx.x], sm[threadIdx.x + s]);
    __syncthreads();
  }
  if (threadIdx.x == 0) atomicMax(slot, __float_as_uint(sm[0]));
}

__global__ void quantize_rows_kernel(const float* __restrict__ src, unsigned char* __restrict__ dst,
                                     int* __restrict__ rowsum, const unsigned* __restrict__ amax_slot) {
  __shared__ int ssum[128];
  const int row = blockIdx.x;
  const int c = threadIdx.x * 4;
  const float s = fmaxf(__uint_as_float(*amax_slot) * (1.0f / 127.0f), 1e-8f);
  const float4 v = *(const float4*)(src + row * IN_K + c);
  const int q0 = (int)fminf(fmaxf(rintf(v.x / s), -128.0f), 127.0f) + 128;
  const int q1 = (int)fminf(fmaxf(rintf(v.y / s), -128.0f), 127.0f) + 128;
  const int q2 = (int)fminf(fmaxf(rintf(v.z / s), -128.0f), 127.0f) + 128;
  const int q3 = (int)fminf(fmaxf(rintf(v.w / s), -128.0f), 127.0f) + 128;
  const unsigned packed = (unsigned)q0 | ((unsigned)q1 << 8) | ((unsigned)q2 << 16) | ((unsigned)q3 << 24);
  *(unsigned*)(dst + row * IN_K + c) = packed;
  ssum[threadIdx.x] = q0 + q1 + q2 + q3;
  __syncthreads();
  for (int t = 64; t > 0; t >>= 1) {
    if (threadIdx.x < (unsigned)t) ssum[threadIdx.x] += ssum[threadIdx.x + t];
    __syncthreads();
  }
  if (threadIdx.x == 0) rowsum[row] = ssum[0];
}

__global__ void residual_kernel(const float* __restrict__ lut, float* __restrict__ R) {
  const int i = blockIdx.x * blockDim.x + threadIdx.x;
  const float4 f = ((const float4*)lut)[i];
  const int idx = i * 4;
  const int a = (idx >> 8) - 128;
  const int b = (idx & 255) - 128;
  float4 r;
  r.x = f.x - (float)(a * b);
  r.y = f.y - (float)(a * (b + 1));
  r.z = f.z - (float)(a * (b + 2));
  r.w = f.w - (float)(a * (b + 3));
  ((float4*)R)[i] = r;
}

__global__ __launch_bounds__(256) void lut_gemm_kernel(
    const unsigned char* __restrict__ xa, const unsigned char* __restrict__ wa,
    const int* __restrict__ Sx, const int* __restrict__ Sw,
    const float* __restrict__ Rg, const float* __restrict__ bias,
    const unsigned* __restrict__ amax, float* __restrict__ out) {
  extern __shared__ char smem[];
  float* Rl = (float*)smem;
  unsigned char* xal = (unsigned char*)(smem + LDS_R_BYTES);
  unsigned char* wal = (unsigned char*)(smem + LDS_R_BYTES + LDS_XA_BYTES);

  const int tid = threadIdx.x;
  const int rowbase = blockIdx.x * 64;
  const int colbase = blockIdx.y * 32;
  const int wave = tid >> 5;
  const int lane = tid & 31;

  if (wave == 0) {
    tdm_load_2d_f32(Rl, Rg, 256u, 256u);
  }
  {
    const char* xg = (const char*)(xa + rowbase * IN_K);
    for (int i = tid; i < 2048; i += 256)
      async_copy_b128(xal + i * 16, xg + i * 16);
    const char* wg = (const char*)(wa + colbase * IN_K);
    for (int i = tid; i < 1024; i += 256)
      async_copy_b128(wal + i * 16, wg + i * 16);
  }
  wait_asynccnt0();
  if (wave == 0) wait_tensorcnt0();
  __syncthreads();

  const int half = lane >> 4;
  const int ln   = lane & 15;
  const int mloc = (wave >> 1) << 4;
  const int nloc = (wave & 1) << 4;

  const unsigned char* arow = xal + (mloc + ln) * IN_K;
  const unsigned char* brow = wal + (nloc + ln) * IN_K;

  v8i acc = {0, 0, 0, 0, 0, 0, 0, 0};
  for (int k0 = 0; k0 < IN_K; k0 += 64) {
    v8i A, Bm;
#pragma unroll
    for (int v = 0; v < 8; ++v) {
      const int ka = ((v >> 2) << 5) + (((v >> 1) & 1) << 4) + ((v & 1) << 2) + (half << 3);
      const int kb = ((v >> 2) << 5) + (((v >> 1) & 1) << 4) + ((v & 1) << 2) + (half << 3);
      A[v]  = *(const int*)(arow + k0 + ka);
      Bm[v] = *(const int*)(brow + k0 + kb);
    }
    acc = __builtin_amdgcn_wmma_i32_16x16x64_iu8(false, A, false, Bm, acc, false, false);
  }

  float res[8] = {0, 0, 0, 0, 0, 0, 0, 0};
  const unsigned char* xrow = xal + (mloc + (half << 3)) * IN_K;
  for (int k = 0; k < IN_K; k += 4) {
    const unsigned bw = *(const unsigned*)(brow + k);
    const int b0 = (int)(bw & 255u), b1 = (int)((bw >> 8) & 255u);
    const int b2 = (int)((bw >> 16) & 255u), b3 = (int)(bw >> 24);
#pragma unroll
    for (int r = 0; r < 8; ++r) {
      const unsigned aw = *(const unsigned*)(xrow + r * IN_K + k);
      res[r] += Rl[(int)((aw        & 255u) << 8) + b0]
             +  Rl[(int)(((aw >>  8) & 255u) << 8) + b1]
             +  Rl[(int)(((aw >> 16) & 255u) << 8) + b2]
             +  Rl[(int)((aw >> 24)          << 8) + b3];
    }
  }

  const float sx = fmaxf(__uint_as_float(amax[0]) * (1.0f / 127.0f), 1e-8f);
  const float sw = fmaxf(__uint_as_float(amax[1]) * (1.0f / 127.0f), 1e-8f);
  const float scale = sx * sw;
  const int ng = colbase + nloc + ln;
  const float bz = bias[ng];
  const int swsum = Sw[ng];
#pragma unroll
  for (int r = 0; r < 8; ++r) {
    const int mg = rowbase + mloc + (half << 3) + r;
    const int dots = acc[r] - 128 * (Sx[mg] + swsum) + 128 * 128 * IN_K;
    out[mg * OUT_N + ng] = ((float)dots + res[r]) * scale + bz;
  }
}

extern "C" void kernel_launch(void* const* d_in, const int* in_sizes, int n_in,
                              void* d_out, int out_size, void* d_ws, size_t ws_size,
                              hipStream_t stream) {
  const float* x    = (const float*)d_in[0];
  const float* w    = (const float*)d_in[1];
  const float* bias = (const float*)d_in[2];
  const float* lut  = (const float*)d_in[3];
  float* out = (float*)d_out;

  char* ws = (char*)d_ws;
  unsigned* amax = (unsigned*)(ws + OFF_AMAX);
  int* Sx = (int*)(ws + OFF_SX);
  int* Sw = (int*)(ws + OFF_SW);
  float* R = (float*)(ws + OFF_R);
  unsigned char* xa = (unsigned char*)(ws + OFF_XA);
  unsigned char* wa = (unsigned char*)(ws + OFF_WA);

  init_kernel<<<1, 32, 0, stream>>>(amax);
  absmax_kernel<<<512, 256, 0, stream>>>(x, B_ROWS * IN_K, amax + 0);
  absmax_kernel<<<256, 256, 0, stream>>>(w, OUT_N * IN_K, amax + 1);
  quantize_rows_kernel<<<B_ROWS, 128, 0, stream>>>(x, xa, Sx, amax + 0);
  quantize_rows_kernel<<<OUT_N, 128, 0, stream>>>(w, wa, Sw, amax + 1);
  residual_kernel<<<64, 256, 0, stream>>>(lut, R);

  static int lds_ok = 0;
  if (!lds_ok) {
    (void)hipFuncSetAttribute(reinterpret_cast<const void*>(lut_gemm_kernel),
                              hipFuncAttributeMaxDynamicSharedMemorySize, LDS_TOTAL);
    lds_ok = 1;
  }
  lut_gemm_kernel<<<dim3(B_ROWS / 64, OUT_N / 32), 256, LDS_TOTAL, stream>>>(
      xa, wa, Sx, Sw, R, bias, amax, out);
}
